// TransformerXLBlock_70643622085250
// MI455X (gfx1250) — matrix operand order repaired and checked by reading, not compiled, not run on hardware
//
#include <hip/hip_runtime.h>
#include <hip/hip_bf16.h>

// ---------------------------------------------------------------------------
// Types / helpers
// ---------------------------------------------------------------------------
typedef __attribute__((ext_vector_type(16))) __bf16 v16bf;
typedef __attribute__((ext_vector_type(8)))  float  v8f;

#define DEVINL __device__ __forceinline__

// Native converts: backend emits hardware bf16 cvt ops (RTNE) instead of
// 3-instruction integer emulation.
DEVINL __bf16 f2bf(float f)  { return (__bf16)f; }
DEVINL float  bf2f(__bf16 b) { return (float)b; }

// ---------------------------------------------------------------------------
// Async global->LDS staging (CDNA5 GLOBAL_LOAD_ASYNC_TO_LDS_B128, ASYNCcnt).
// Builtin signature (from compiler diagnostic): param 1 is
// `int __attribute__((vector_size(16))) addrspace(1)*` (global src),
// param 2 the LDS dst, then constant offset + cpol.
// Guarded: falls back to a synchronous b128 copy if the builtin is absent.
// ---------------------------------------------------------------------------
#if __has_builtin(__builtin_amdgcn_global_load_async_to_lds_b128)
#define HAS_ASYNC_LDS 1
typedef int v4i_vs __attribute__((vector_size(16)));
typedef __attribute__((address_space(1))) v4i_vs gv4i;
typedef __attribute__((address_space(3))) v4i_vs lv4i;
#endif

DEVINL void stage16(const __bf16* g, __bf16* l) {
#ifdef HAS_ASYNC_LDS
  __builtin_amdgcn_global_load_async_to_lds_b128((gv4i*)g, (lv4i*)l, 0, 0);
#else
  *(uint4*)l = *(const uint4*)g;
#endif
}
DEVINL void async_fence() {
#ifdef HAS_ASYNC_LDS
#if __has_builtin(__builtin_amdgcn_s_wait_asynccnt)
  __builtin_amdgcn_s_wait_asynccnt(0);
#else
  asm volatile("s_wait_asynccnt 0" ::: "memory");
#endif
#endif
}

// ---------------------------------------------------------------------------
// WMMA fragment loaders (ISA 7.12.2 layouts)
// ---------------------------------------------------------------------------
// A-fragment (16x32 bf16): lane m=lane&15 holds row m; K elements
// {h*8 .. h*8+7, 16+h*8 .. 16+h*8+7} with h = lane>>4.
DEVINL v16bf load_frag_a(const __bf16* tile, int ld, int lane) {
  const int m = lane & 15, h = lane >> 4;
  const __bf16* p = tile + m * ld + h * 8;
  union { v16bf v; uint4 u[2]; } f;
  f.u[0] = *(const uint4*)(p);
  f.u[1] = *(const uint4*)(p + 16);
  return f.v;
}
// B-fragment (32x16 bf16): lane n=lane&15 holds column n; K elements
// {h*8 .. h*8+7, 16+h*8 .. 16+h*8+7}. Tile must be stored as [n][k] (i.e. B^T) in LDS.
DEVINL v16bf load_frag_b(const __bf16* tile, int ld, int lane) {
  const int n = lane & 15, h = lane >> 4;
  const __bf16* p = tile + n * ld + h * 8;
  union { v16bf v; uint4 u[2]; } f;
  f.u[0] = *(const uint4*)(p);
  f.u[1] = *(const uint4*)(p + 16);
  return f.v;
}
DEVINL v8f wmma_bf16(v16bf a, v16bf b, v8f c) {
  return __builtin_amdgcn_wmma_f32_16x16x32_bf16(false, a, false, b, (short)0, c,
                                                 false, false);
}

DEVINL void cvt4(const float* s, __bf16* d) {
  float4 v = *(const float4*)s;
  d[0] = f2bf(v.x); d[1] = f2bf(v.y); d[2] = f2bf(v.z); d[3] = f2bf(v.w);
}
DEVINL void store_out(float* p, float v)  { *p = v; }
DEVINL void store_out(__bf16* p, float v) { *p = f2bf(v); }

// Tile staging overloads for the GEMM A operand.
// f32 source: load float4, convert to bf16 (VALU cvt), store to LDS.
DEVINL void stage_tileA(const float* A, __bf16* sA, int tid, int row0, int K,
                        int k0, int ldt) {
#pragma unroll
  for (int it = 0; it < 4; it++) {
    int idx = tid + it * 256;
    int r = idx >> 3, c4 = (idx & 7) << 2;
    cvt4(A + (size_t)(row0 + r) * K + (k0 + c4), sA + r * ldt + c4);
  }
}
// bf16 source: pure copy -> async b128 global->LDS.
DEVINL void stage_tileA(const __bf16* A, __bf16* sA, int tid, int row0, int K,
                        int k0, int ldt) {
#pragma unroll
  for (int it = 0; it < 2; it++) {
    int idx = tid + it * 256;
    int r = idx >> 2, c8 = (idx & 3) << 3;
    stage16(A + (size_t)(row0 + r) * K + (k0 + c8), sA + r * ldt + c8);
  }
}

// ---------------------------------------------------------------------------
// Generic tiled WMMA GEMM: C[M,N] = epilogue(A[M,K] * B[K,N])
// 256 threads = 8 waves, tile 128x128x32, wave tile 64x32 (4x2 WMMA tiles).
// EPI: 0 = none, 1 = +bias, 2 = +bias then ReLU
// ---------------------------------------------------------------------------
template <int EPI, typename TOUT, typename TA>
__global__ __launch_bounds__(256)
void gemm_wmma_kernel(const TA* __restrict__ A, const float* __restrict__ B,
                      const float* __restrict__ bias, TOUT* __restrict__ C,
                      int M, int N, int K) {
  constexpr int BM = 128, BN = 128, BK = 32, LDT = BK + 8;  // 80B rows
  __shared__ alignas(16) __bf16 sA[BM * LDT];
  __shared__ alignas(16) __bf16 sB[BN * LDT];  // B^T tile: [n][k]
  const int tid = threadIdx.x, lane = tid & 31, wave = tid >> 5;
  const int wm = wave >> 2, wn = wave & 3;
  const int row0 = blockIdx.y * BM, col0 = blockIdx.x * BN;

  v8f acc[4][2];
#pragma unroll
  for (int mt = 0; mt < 4; mt++)
#pragma unroll
    for (int nt = 0; nt < 2; nt++)
#pragma unroll
      for (int e = 0; e < 8; e++) acc[mt][nt][e] = 0.f;

  for (int k0 = 0; k0 < K; k0 += BK) {
    __syncthreads();
    stage_tileA(A, sA, tid, row0, K, k0, LDT);
    // Stage B^T (32x128 -> [n][k]), coalesced f32 reads, scattered bf16 store.
#pragma unroll
    for (int it = 0; it < 16; it++) {
      int idx = tid + it * 256;
      int n = idx & (BN - 1), kk = idx >> 7;
      sB[n * LDT + kk] = f2bf(B[(size_t)(k0 + kk) * N + col0 + n]);
    }
    if (k0 + BK < K) {  // prefetch next tiles -> global_prefetch_b8
      __builtin_prefetch(A + (size_t)(row0 + (tid >> 1)) * K + (k0 + BK) + (tid & 1) * 16, 0, 1);
      __builtin_prefetch(B + (size_t)(k0 + BK + (tid >> 3)) * N + col0 + (tid & 7) * 16, 0, 1);
    }
    async_fence();
    __syncthreads();

    v16bf af[4], bfr[2];
#pragma unroll
    for (int mt = 0; mt < 4; mt++)
      af[mt] = load_frag_a(sA + (wm * 64 + mt * 16) * LDT, LDT, lane);
#pragma unroll
    for (int nt = 0; nt < 2; nt++)
      bfr[nt] = load_frag_b(sB + (wn * 32 + nt * 16) * LDT, LDT, lane);
#pragma unroll
    for (int mt = 0; mt < 4; mt++)
#pragma unroll
      for (int nt = 0; nt < 2; nt++)
        acc[mt][nt] = wmma_bf16(af[mt], bfr[nt], acc[mt][nt]);
  }

  const int half = lane >> 4, nl = lane & 15;
#pragma unroll
  for (int mt = 0; mt < 4; mt++)
#pragma unroll
    for (int nt = 0; nt < 2; nt++)
#pragma unroll
      for (int r = 0; r < 8; r++) {
        int row = row0 + wm * 64 + mt * 16 + half * 8 + r;
        int col = col0 + wn * 32 + nt * 16 + nl;
        float v = acc[mt][nt][r];
        if (EPI >= 1) v += bias[col];
        if (EPI == 2) v = fmaxf(v, 0.f);
        store_out(C + (size_t)row * N + col, v);
      }
}

// ---------------------------------------------------------------------------
// Fused flash attention with Transformer-XL relative shift.
// rel_shift identity: shifted[i,j] = pos_raw[i, j - i + S].
// Block = (b, n, 64 queries), 128 threads = 4 waves, each wave owns 16 rows.
// Per 64-key tile: Sc = Qc*K^T (WMMA), Sp = Qp*R[dbase..dbase+126]^T (WMMA),
// score[i,j] = (Sc + Sp[i, j-i+63]) / 8, online softmax, O += P*V (WMMA).
// K and R tiles are staged with async b128 global->LDS copies.
// ---------------------------------------------------------------------------
__global__ __launch_bounds__(128)
void attn_flash_kernel(const __bf16* __restrict__ qb, const __bf16* __restrict__ kb,
                       const __bf16* __restrict__ vb, const __bf16* __restrict__ rb,
                       const float* __restrict__ cbias, const float* __restrict__ pbias,
                       __bf16* __restrict__ attn) {
  constexpr int S = 1024, NH = 16, H = 64, LDT = 72, LSP = 136;
  constexpr float LOG2E = 1.4426950408889634f;
  __shared__ alignas(16) __bf16 sQc[64 * LDT];
  __shared__ alignas(16) __bf16 sQp[64 * LDT];
  __shared__ alignas(16) __bf16 sK[64 * LDT];
  __shared__ alignas(16) __bf16 sR[128 * LDT];       // also hosts V^T + P later
  __shared__ alignas(16) __bf16 sSp[4 * 16 * LSP];   // per-wave Sp scratch
  __bf16* sV = sR;              // rows 0..63  : V^T tile [h][j]
  __bf16* sP = sR + 64 * LDT;   // rows 64..127: P tile [i][j]

  const int tid = threadIdx.x, lane = tid & 31, wave = tid >> 5;
  const int half = lane >> 4, nl = lane & 15;
  const int i0 = blockIdx.x * 64;
  const int n = blockIdx.y, b = blockIdx.z;
  const size_t strideS = (size_t)NH * H;
  const size_t baseq = (size_t)b * S * strideS + (size_t)n * H;
  const size_t baser = (size_t)b * 2 * S * strideS + (size_t)n * H;

  // Stage Qc = q + content_bias, Qp = q + positional_bias (8-wide vector ops)
#pragma unroll
  for (int it = 0; it < 4; it++) {
    int idx = tid + it * 128;
    int rr = idx >> 3, c8 = (idx & 7) << 3;
    union { uint4 u; __bf16 h[8]; } qv, oc, op;
    qv.u = *(const uint4*)(qb + baseq + (size_t)(i0 + rr) * strideS + c8);
#pragma unroll
    for (int e = 0; e < 8; e++) {
      float q = bf2f(qv.h[e]);
      oc.h[e] = f2bf(q + cbias[n * H + c8 + e]);
      op.h[e] = f2bf(q + pbias[n * H + c8 + e]);
    }
    *(uint4*)(sQc + rr * LDT + c8) = oc.u;
    *(uint4*)(sQp + rr * LDT + c8) = op.u;
  }

  float mrun[8], lrun[8];
  v8f o[4];
#pragma unroll
  for (int r = 0; r < 8; r++) { mrun[r] = -3.0e38f; lrun[r] = 0.f; }
#pragma unroll
  for (int t = 0; t < 4; t++)
#pragma unroll
    for (int e = 0; e < 8; e++) o[t][e] = 0.f;

  __bf16* mySp = sSp + wave * 16 * LSP;

  for (int j0 = 0; j0 < S; j0 += 64) {
    const int dbase = S + j0 - i0 - 63;  // in [1, 2S-127]
    __syncthreads();  // prior-iter PV reads done; safe to restage K / R
    // K tile: 64x64 bf16, async b128 chunks.
#pragma unroll
    for (int it = 0; it < 4; it++) {
      int idx = tid + it * 128;
      int rr = idx >> 3, c8 = (idx & 7) << 3;
      stage16(kb + baseq + (size_t)(j0 + rr) * strideS + c8, sK + rr * LDT + c8);
    }
    // R tile: 127x64 bf16 (row 127 zero), async b128 chunks.
#pragma unroll
    for (int it = 0; it < 8; it++) {
      int idx = tid + it * 128;
      int rr = idx >> 3, c8 = (idx & 7) << 3;
      if (rr < 127) {
        stage16(rb + baser + (size_t)(dbase + rr) * strideS + c8, sR + rr * LDT + c8);
      } else {
        uint4 z = {0u, 0u, 0u, 0u};
        *(uint4*)(sR + rr * LDT + c8) = z;
      }
    }
    async_fence();
    __syncthreads();

    // Content 64x64 and position 64x128 scores for this wave's 16 rows.
    v8f sc[4], sp[8];
#pragma unroll
    for (int t = 0; t < 4; t++)
#pragma unroll
      for (int e = 0; e < 8; e++) sc[t][e] = 0.f;
#pragma unroll
    for (int dt = 0; dt < 8; dt++)
#pragma unroll
      for (int e = 0; e < 8; e++) sp[dt][e] = 0.f;

#pragma unroll
    for (int kk = 0; kk < 2; kk++) {
      v16bf aq = load_frag_a(sQc + wave * 16 * LDT + kk * 32, LDT, lane);
#pragma unroll
      for (int t = 0; t < 4; t++) {
        v16bf bk = load_frag_b(sK + t * 16 * LDT + kk * 32, LDT, lane);
        sc[t] = wmma_bf16(aq, bk, sc[t]);
      }
      v16bf ap = load_frag_a(sQp + wave * 16 * LDT + kk * 32, LDT, lane);
#pragma unroll
      for (int dt = 0; dt < 8; dt++) {
        v16bf br = load_frag_b(sR + dt * 16 * LDT + kk * 32, LDT, lane);
        sp[dt] = wmma_bf16(ap, br, sp[dt]);
      }
    }

    // Spill Sp to per-wave LDS, then shear-gather: rd = j - i + 63.
#pragma unroll
    for (int dt = 0; dt < 8; dt++)
#pragma unroll
      for (int r = 0; r < 8; r++)
        mySp[(half * 8 + r) * LSP + dt * 16 + nl] = f2bf(sp[dt][r]);

    v8f st[4];
#pragma unroll
    for (int t = 0; t < 4; t++)
#pragma unroll
      for (int r = 0; r < 8; r++) {
        int il = wave * 16 + half * 8 + r;       // local query row (0..63)
        int rd = t * 16 + nl - il + 63;          // 0..126
        float pv = bf2f(mySp[(half * 8 + r) * LSP + rd]);
        st[t][r] = (sc[t][r] + pv) * 0.125f;     // 1/sqrt(64)
      }

    // Online softmax: rows live as element r across 16 lanes of each half.
#pragma unroll
    for (int r = 0; r < 8; r++) {
      float mx = mrun[r];
#pragma unroll
      for (int t = 0; t < 4; t++) mx = fmaxf(mx, st[t][r]);
#pragma unroll
      for (int off = 1; off < 16; off <<= 1) mx = fmaxf(mx, __shfl_xor(mx, off, 32));
      float corr = exp2f((mrun[r] - mx) * LOG2E);
      float rs = 0.f;
#pragma unroll
      for (int t = 0; t < 4; t++) {
        float p = exp2f((st[t][r] - mx) * LOG2E);
        st[t][r] = p;
        rs += p;
      }
#pragma unroll
      for (int off = 1; off < 16; off <<= 1) rs += __shfl_xor(rs, off, 32);
      lrun[r] = lrun[r] * corr + rs;
      mrun[r] = mx;
#pragma unroll
      for (int t = 0; t < 4; t++) o[t][r] = o[t][r] * corr;
    }

    __syncthreads();  // all waves done reading K/R; overlay V^T + P on sR
    // V^T staging: coalesced 4B global reads, two 2B LDS stores (transpose).
#pragma unroll
    for (int it = 0; it < 16; it++) {
      int idx = tid + it * 128;
      int jj = idx >> 5, h2 = (idx & 31) << 1;
      union { unsigned u; __bf16 h[2]; } vv;
      vv.u = *(const unsigned*)(vb + baseq + (size_t)(j0 + jj) * strideS + h2);
      sV[h2 * LDT + jj] = vv.h[0];
      sV[(h2 + 1) * LDT + jj] = vv.h[1];
    }
#pragma unroll
    for (int t = 0; t < 4; t++)
#pragma unroll
      for (int r = 0; r < 8; r++)
        sP[(wave * 16 + half * 8 + r) * LDT + t * 16 + nl] = f2bf(st[t][r]);
    __syncthreads();

    // O += P * V  (K-dim = 64 keys = 2 WMMA steps)
#pragma unroll
    for (int kk = 0; kk < 2; kk++) {
      v16bf apv = load_frag_a(sP + wave * 16 * LDT + kk * 32, LDT, lane);
#pragma unroll
      for (int t = 0; t < 4; t++) {
        v16bf bv = load_frag_b(sV + t * 16 * LDT + kk * 32, LDT, lane);
        o[t] = wmma_bf16(apv, bv, o[t]);
      }
    }
  }

  // Normalize and write attention output [b, s, n, h] in bf16.
#pragma unroll
  for (int t = 0; t < 4; t++)
#pragma unroll
    for (int r = 0; r < 8; r++) {
      int rowl = wave * 16 + half * 8 + r;
      float vo = o[t][r] / lrun[r];
      attn[baseq + (size_t)(i0 + rowl) * strideS + t * 16 + nl] = f2bf(vo);
    }
}

// ---------------------------------------------------------------------------
// LayerNorm over last dim: out = LN(y + resid) * g + b   (one block per row)
// ---------------------------------------------------------------------------
__global__ __launch_bounds__(256)
void ln_kernel(const float* __restrict__ y, const float* __restrict__ resid,
               const float* __restrict__ g, const float* __restrict__ bb,
               float* __restrict__ out, int E) {
  __shared__ float s1[256], s2[256];
  const int row = blockIdx.x, tid = threadIdx.x;
  const float* py = y + (size_t)row * E;
  const float* pr = resid + (size_t)row * E;
  float a = 0.f, q = 0.f;
  for (int c = tid; c < E; c += 256) {
    float v = py[c] + pr[c];
    a += v;
    q += v * v;
  }
  s1[tid] = a; s2[tid] = q;
  __syncthreads();
  for (int off = 128; off > 0; off >>= 1) {
    if (tid < off) { s1[tid] += s1[tid + off]; s2[tid] += s2[tid + off]; }
    __syncthreads();
  }
  float mu = s1[0] / (float)E;
  float var = fmaxf(s2[0] / (float)E - mu * mu, 0.f);
  float rstd = rsqrtf(var + 1e-12f);
  for (int c = tid; c < E; c += 256) {
    float v = py[c] + pr[c];
    out[(size_t)row * E + c] = (v - mu) * rstd * g[c] + bb[c];
  }
}

// ---------------------------------------------------------------------------
// Launch
// ---------------------------------------------------------------------------
extern "C" void kernel_launch(void* const* d_in, const int* in_sizes, int n_in,
                              void* d_out, int out_size, void* d_ws, size_t ws_size,
                              hipStream_t stream) {
  (void)in_sizes; (void)n_in; (void)out_size; (void)ws_size;
  constexpr int Bb = 4, S = 1024, E = 1024, Nh = 16, F = 4096;
  constexpr int M = Bb * S;  // 4096 rows

  const float* x    = (const float*)d_in[0];
  const float* cb   = (const float*)d_in[1];
  const float* pb   = (const float*)d_in[2];
  const float* rp   = (const float*)d_in[3];
  const float* Wq   = (const float*)d_in[4];
  const float* Wk   = (const float*)d_in[5];
  const float* Wv   = (const float*)d_in[6];
  const float* Wr   = (const float*)d_in[7];
  const float* Wo   = (const float*)d_in[8];
  const float* g1   = (const float*)d_in[9];
  const float* b1   = (const float*)d_in[10];
  const float* Wi   = (const float*)d_in[11];
  const float* bi   = (const float*)d_in[12];
  const float* Wout = (const float*)d_in[13];
  const float* bo   = (const float*)d_in[14];
  const float* g2   = (const float*)d_in[15];
  const float* b2   = (const float*)d_in[16];
  float* out = (float*)d_out;

  char* ws = (char*)d_ws;
  const size_t MB = 1024ull * 1024ull;
  __bf16* qb    = (__bf16*)(ws + 0 * MB);     // 8 MB   [B,S,N,H] bf16
  __bf16* kb    = (__bf16*)(ws + 8 * MB);     // 8 MB
  __bf16* vb    = (__bf16*)(ws + 16 * MB);    // 8 MB
  __bf16* rb    = (__bf16*)(ws + 24 * MB);    // 16 MB  [B,2S,N,H] bf16
  __bf16* attnb = (__bf16*)(ws + 40 * MB);    // 8 MB   [B,S,N,H] bf16
  float*  yb    = (float*)(ws + 48 * MB);     // 16 MB  attn proj out
  float*  hb    = (float*)(ws + 64 * MB);     // 16 MB  post-LN1
  float*  inner = (float*)(ws + 80 * MB);     // 64 MB  FFN hidden
  float*  ffn   = (float*)(ws + 144 * MB);    // 16 MB  FFN out

  dim3 blk(256);

  // Projections: q/k/v = x * W, r = relpos * Wr  (bf16 outputs)
  gemm_wmma_kernel<0, __bf16, float>
      <<<dim3(E / 128, M / 128), blk, 0, stream>>>(x, Wq, nullptr, qb, M, E, E);
  gemm_wmma_kernel<0, __bf16, float>
      <<<dim3(E / 128, M / 128), blk, 0, stream>>>(x, Wk, nullptr, kb, M, E, E);
  gemm_wmma_kernel<0, __bf16, float>
      <<<dim3(E / 128, M / 128), blk, 0, stream>>>(x, Wv, nullptr, vb, M, E, E);
  gemm_wmma_kernel<0, __bf16, float>
      <<<dim3(E / 128, (2 * M) / 128), blk, 0, stream>>>(rp, Wr, nullptr, rb,
                                                         2 * M, E, E);

  // Fused relative-position flash attention.
  attn_flash_kernel<<<dim3(S / 64, Nh, Bb), dim3(128), 0, stream>>>(
      qb, kb, vb, rb, cb, pb, attnb);

  // Output projection: y = attn * Wo   ([N*H, E] row-major)  (async A staging)
  gemm_wmma_kernel<0, float, __bf16>
      <<<dim3(E / 128, M / 128), blk, 0, stream>>>(attnb, Wo, nullptr, yb, M, E, E);

  // h = LN(y + x)
  ln_kernel<<<dim3(M), blk, 0, stream>>>(yb, x, g1, b1, hb, E);

  // inner = relu(h * Wi + bi)
  gemm_wmma_kernel<2, float, float>
      <<<dim3(F / 128, M / 128), blk, 0, stream>>>(hb, Wi, bi, inner, M, F, E);

  // ffn = inner * Wout + bout
  gemm_wmma_kernel<1, float, float>
      <<<dim3(E / 128, M / 128), blk, 0, stream>>>(inner, Wout, bo, ffn, M, E, F);

  // out = LN(ffn + h)
  ln_kernel<<<dim3(M), blk, 0, stream>>>(ffn, hb, g2, b2, out, E);
}
